// DMCell_10222022164698
// MI455X (gfx1250) — matrix operand order repaired and checked by reading, not compiled, not run on hardware
//
#include <hip/hip_runtime.h>
#include <hip/hip_bf16.h>

// ---------------------------------------------------------------------------
// CDNA5 / gfx1250 FORCE-RLS step: bf16 WMMA GEMMs, f32 accumulation,
// 128x128x32 block tiles, 32x64 per-wave tiles (8 WMMA / K-step / wave),
// double-buffered LDS staging with f32->bf16 conversion in flight.
// ---------------------------------------------------------------------------

typedef __bf16 bf16;
typedef __attribute__((ext_vector_type(16))) __bf16 v16bf;
typedef __attribute__((ext_vector_type(8)))  float  v8f;

#define B_DIM 8192
#define I_DIM 2048
#define H_DIM 1024

#define TM 128
#define TN 128
#define TK 32
#define LDS_NT  40   // k-stride (bf16) for [row][k] tiles (32 + 8 pad)
#define LDS_TNS 136  // col-stride (bf16) for [k][col] tiles (128 + 8 pad)

#define A_WW    270.0f
#define BB_WW   108.0f
#define D_WW    0.154f
#define ALPHA_C 0.01f
#define GAMMA_C 0.1f

__device__ __forceinline__ float log_act_f(float x) {
  float z  = D_WW * (A_WW * x - BB_WW);
  float az = __builtin_fabsf(z);
  float zs = (az < 1e-6f) ? 1.0f : z;
  float v  = zs / (D_WW * (1.0f - __expf(-zs)));
  return (az < 1e-6f) ? (1.0f / D_WW) : v;
}

// ---- global(f32) -> LDS(bf16) staging --------------------------------------
// NT operand: row-major [rows][K], K contiguous. Tile: 128 rows x 32 k.
__device__ __forceinline__ void stage_nt(const float* __restrict__ g, int ld,
                                         int r0, int k0, bf16* lds, int tid) {
  int c4  = (tid & 7) << 2;   // k offset 0..28
  int rr0 = tid >> 3;         // 0..31
#pragma unroll
  for (int rep = 0; rep < 4; ++rep) {
    int rr = rr0 + rep * 32;
    const float4 v = *(const float4*)(g + (size_t)(r0 + rr) * ld + (k0 + c4));
    bf16* d = lds + rr * LDS_NT + c4;
    d[0] = (bf16)v.x; d[1] = (bf16)v.y; d[2] = (bf16)v.z; d[3] = (bf16)v.w;
  }
}

// TN operand: row-major [K][cols], cols contiguous. Tile: 32 k x 128 cols.
// Optional per-k row scale (applies 1/(1+rPr) -> materializes k on the fly).
__device__ __forceinline__ void stage_tn(const float* __restrict__ g, int ld,
                                         int k0, int c0, bf16* lds,
                                         const float* __restrict__ scale, int tid) {
  int c4  = (tid & 31) << 2;  // col offset 0..124
  int kr0 = tid >> 5;         // 0..7
#pragma unroll
  for (int rep = 0; rep < 4; ++rep) {
    int kr = kr0 + rep * 8;
    const float4 v = *(const float4*)(g + (size_t)(k0 + kr) * ld + (c0 + c4));
    float sc = scale ? scale[k0 + kr] : 1.0f;
    bf16* d = lds + kr * LDS_TNS + c4;
    d[0] = (bf16)(v.x * sc); d[1] = (bf16)(v.y * sc);
    d[2] = (bf16)(v.z * sc); d[3] = (bf16)(v.w * sc);
  }
}

// ---- WMMA fragment gathers (ISA 7.12.2 layouts, wave32) --------------------
// A 16x32 bf16: lane m=lane&15, half=lane>>4; element e: k = e + (e>=8?8:0) + 8*half
__device__ __forceinline__ v16bf frag_a_nt(const bf16* lA, int m, int half) {
  v16bf a;
#pragma unroll
  for (int e = 0; e < 16; ++e) {
    int k = e + ((e >= 8) ? 8 : 0) + 8 * half;
    a[e] = lA[m * LDS_NT + k];
  }
  return a;
}
// B 32x16 bf16: lane n=lane&15; element e: k = e + (e>=8?8:0) + 8*half (same as A).
__device__ __forceinline__ v16bf frag_b_nt(const bf16* lB, int n, int half) {
  v16bf b;
#pragma unroll
  for (int e = 0; e < 16; ++e) b[e] = lB[n * LDS_NT + e + ((e >= 8) ? 8 : 0) + 8 * half];
  return b;
}
__device__ __forceinline__ v16bf frag_a_tn(const bf16* lA, int m, int half) {
  v16bf a;
#pragma unroll
  for (int e = 0; e < 16; ++e) {
    int k = e + ((e >= 8) ? 8 : 0) + 8 * half;
    a[e] = lA[k * LDS_TNS + m];
  }
  return a;
}
__device__ __forceinline__ v16bf frag_b_tn(const bf16* lB, int n, int half) {
  v16bf b;
#pragma unroll
  for (int e = 0; e < 16; ++e) b[e] = lB[(e + ((e >= 8) ? 8 : 0) + 8 * half) * LDS_TNS + n];
  return b;
}

#define WMMA_BF16(ACC, AF, BF)                                                  \
  (ACC) = __builtin_amdgcn_wmma_f32_16x16x32_bf16(false, (AF), false, (BF),     \
                                                  (short)0, (ACC), false, false)

// Per-wave NT compute step: 2 A-frags x 4 B-frags -> 8 WMMAs.
__device__ __forceinline__ void compute_nt(const bf16* lA, const bf16* lB,
                                           int wm, int wn, int ncol, int half,
                                           v8f acc[2][4]) {
  v16bf af[2], bfm[4];
#pragma unroll
  for (int mi = 0; mi < 2; ++mi)
    af[mi] = frag_a_nt(lA, wm * 32 + mi * 16 + ncol, half);
#pragma unroll
  for (int ni = 0; ni < 4; ++ni)
    bfm[ni] = frag_b_nt(lB, wn * 64 + ni * 16 + ncol, half);
#pragma unroll
  for (int mi = 0; mi < 2; ++mi)
#pragma unroll
    for (int ni = 0; ni < 4; ++ni) WMMA_BF16(acc[mi][ni], af[mi], bfm[ni]);
}

__device__ __forceinline__ void compute_tn(const bf16* lA, const bf16* lB,
                                           int wm, int wn, int ncol, int half,
                                           v8f acc[2][4]) {
  v16bf af[2], bfm[4];
#pragma unroll
  for (int mi = 0; mi < 2; ++mi)
    af[mi] = frag_a_tn(lA, wm * 32 + mi * 16 + ncol, half);
#pragma unroll
  for (int ni = 0; ni < 4; ++ni)
    bfm[ni] = frag_b_tn(lB, wn * 64 + ni * 16 + ncol, half);
#pragma unroll
  for (int mi = 0; mi < 2; ++mi)
#pragma unroll
    for (int ni = 0; ni < 4; ++ni) WMMA_BF16(acc[mi][ni], af[mi], bfm[ni]);
}

// ---------------------------------------------------------------------------
// Kernel 1: rx = x@win^T + s@wr^T ; err, r=log_act(rx), s_new fused epilogue.
// Grid: (B/128, H/128). 8 waves in 4(M)x2(N); each wave owns 32x64 of C.
// Double-buffered LDS; phase-2 first tile pre-staged during phase-1 tail.
// ---------------------------------------------------------------------------
__global__ __launch_bounds__(256)
void rx_fused_kernel(const float* __restrict__ x, const float* __restrict__ win,
                     const float* __restrict__ s, const float* __restrict__ wr,
                     const float* __restrict__ y,
                     float* __restrict__ out_err, float* __restrict__ out_r,
                     float* __restrict__ out_sn) {
  __shared__ bf16 lA[2][TM * LDS_NT];
  __shared__ bf16 lB[2][TN * LDS_NT];
  const int tid  = threadIdx.x;
  const int wave = tid >> 5, lane = tid & 31;
  const int half = lane >> 4, ncol = lane & 15;
  const int wm = wave & 3, wn = wave >> 2;
  const int m0 = blockIdx.x * TM;   // batch rows
  const int n0 = blockIdx.y * TN;   // H cols
  v8f acc[2][4] = {};

  // prologue: stage phase-1 first tile
  stage_nt(x,   I_DIM, m0, 0, lA[0], tid);
  stage_nt(win, I_DIM, n0, 0, lB[0], tid);
  int buf = 0;

  // phase 1: contract over I with (x, win)
  for (int k0 = 0; k0 < I_DIM; k0 += TK) {
    __syncthreads();
    if (k0 + TK < I_DIM) {
      stage_nt(x,   I_DIM, m0, k0 + TK, lA[buf ^ 1], tid);
      stage_nt(win, I_DIM, n0, k0 + TK, lB[buf ^ 1], tid);
      if (k0 + 2 * TK < I_DIM) {
        __builtin_prefetch(x   + (size_t)(m0 + (tid >> 3)) * I_DIM + k0 + 2 * TK, 0, 3);
        __builtin_prefetch(win + (size_t)(n0 + (tid >> 3)) * I_DIM + k0 + 2 * TK, 0, 3);
      }
    } else {  // pre-stage phase-2 first tile: pipeline stays full across phases
      stage_nt(s,  H_DIM, m0, 0, lA[buf ^ 1], tid);
      stage_nt(wr, H_DIM, n0, 0, lB[buf ^ 1], tid);
    }
    compute_nt(lA[buf], lB[buf], wm, wn, ncol, half, acc);
    buf ^= 1;
  }
  // phase 2: contract over H with (s, wr)
  for (int k0 = 0; k0 < H_DIM; k0 += TK) {
    __syncthreads();
    if (k0 + TK < H_DIM) {
      stage_nt(s,  H_DIM, m0, k0 + TK, lA[buf ^ 1], tid);
      stage_nt(wr, H_DIM, n0, k0 + TK, lB[buf ^ 1], tid);
    }
    compute_nt(lA[buf], lB[buf], wm, wn, ncol, half, acc);
    buf ^= 1;
  }
  // fused epilogue
#pragma unroll
  for (int mi = 0; mi < 2; ++mi)
#pragma unroll
    for (int ni = 0; ni < 4; ++ni) {
      int gn = n0 + wn * 64 + ni * 16 + ncol;
#pragma unroll
      for (int v = 0; v < 8; ++v) {
        int gm = m0 + wm * 32 + mi * 16 + v + 8 * half;  // C: m = vgpr + 8*half
        size_t idx = (size_t)gm * H_DIM + gn;
        float rx  = acc[mi][ni][v];                      // I0 == 0
        float err = rx - y[idx];
        float r   = log_act_f(rx);
        float sv  = s[idx];
        out_err[idx] = err;
        out_r[idx]   = r;
        out_sn[idx]  = sv + ALPHA_C * (-sv + (1.0f - sv) * GAMMA_C * r);
      }
    }
}

// ---------------------------------------------------------------------------
// Kernel 2: generic NT GEMM  C[M,N] = A[M,K] * B[N,K]^T  (f32 out).
// Used for k_fenmu = x @ P^T. Grid: (M/128, N/128).
// ---------------------------------------------------------------------------
__global__ __launch_bounds__(256)
void gemm_nt_kernel(const float* __restrict__ A, const float* __restrict__ Bm,
                    float* __restrict__ C, int lda, int ldb, int ldc, int Kdim) {
  __shared__ bf16 lA[2][TM * LDS_NT];
  __shared__ bf16 lB[2][TN * LDS_NT];
  const int tid  = threadIdx.x;
  const int wave = tid >> 5, lane = tid & 31;
  const int half = lane >> 4, ncol = lane & 15;
  const int wm = wave & 3, wn = wave >> 2;
  const int m0 = blockIdx.x * TM;
  const int n0 = blockIdx.y * TN;
  v8f acc[2][4] = {};

  stage_nt(A,  lda, m0, 0, lA[0], tid);
  stage_nt(Bm, ldb, n0, 0, lB[0], tid);
  int buf = 0;

  for (int k0 = 0; k0 < Kdim; k0 += TK) {
    __syncthreads();
    if (k0 + TK < Kdim) {
      stage_nt(A,  lda, m0, k0 + TK, lA[buf ^ 1], tid);
      stage_nt(Bm, ldb, n0, k0 + TK, lB[buf ^ 1], tid);
      if (k0 + 2 * TK < Kdim) {
        __builtin_prefetch(A  + (size_t)(m0 + (tid >> 3)) * lda + k0 + 2 * TK, 0, 3);
        __builtin_prefetch(Bm + (size_t)(n0 + (tid >> 3)) * ldb + k0 + 2 * TK, 0, 3);
      }
    }
    compute_nt(lA[buf], lB[buf], wm, wn, ncol, half, acc);
    buf ^= 1;
  }
#pragma unroll
  for (int mi = 0; mi < 2; ++mi)
#pragma unroll
    for (int ni = 0; ni < 4; ++ni) {
      int gn = n0 + wn * 64 + ni * 16 + ncol;
#pragma unroll
      for (int v = 0; v < 8; ++v) {
        int gm = m0 + wm * 32 + mi * 16 + v + 8 * half;
        C[(size_t)gm * ldc + gn] = acc[mi][ni][v];
      }
    }
}

// ---------------------------------------------------------------------------
// Kernel 3: scale[b] = 1 / (1 + sum_i kf[b,i]*x[b,i]).  Grid: B blocks.
// ---------------------------------------------------------------------------
__global__ __launch_bounds__(256)
void rowscale_kernel(const float* __restrict__ kf, const float* __restrict__ x,
                     float* __restrict__ scale) {
  __shared__ float red[256];
  const int b = blockIdx.x;
  const float* kr = kf + (size_t)b * I_DIM;
  const float* xr = x  + (size_t)b * I_DIM;
  float p = 0.0f;
  for (int i = threadIdx.x; i < I_DIM; i += 256) p += kr[i] * xr[i];
  red[threadIdx.x] = p;
  __syncthreads();
  for (int off = 128; off > 0; off >>= 1) {
    if (threadIdx.x < off) red[threadIdx.x] += red[threadIdx.x + off];
    __syncthreads();
  }
  if (threadIdx.x == 0) scale[b] = 1.0f / (1.0f + red[0]);
}

// ---------------------------------------------------------------------------
// Kernel 4/5: TN rank-update GEMM, K = batch dimension:
//   out[m,n] = base[m,n] + alpha * sum_b (A[b,m]*sA[b]) * (B[b,n]*sB[b])
// Per-b scale applied during LDS staging (materializes k = kf/(1+rPr) on fly).
// Grid: (M/128, N/128).
// ---------------------------------------------------------------------------
__global__ __launch_bounds__(256)
void update_tn_kernel(const float* __restrict__ Aop, int lda,
                      const float* __restrict__ Bop, int ldb,
                      const float* __restrict__ base, float* __restrict__ out,
                      int ldc, const float* __restrict__ scale,
                      int scale_a, int scale_b, int Kdim, float alpha) {
  __shared__ bf16 lA[2][TK * LDS_TNS];
  __shared__ bf16 lB[2][TK * LDS_TNS];
  const int tid  = threadIdx.x;
  const int wave = tid >> 5, lane = tid & 31;
  const int half = lane >> 4, ncol = lane & 15;
  const int wm = wave & 3, wn = wave >> 2;
  const int m0 = blockIdx.x * TM;
  const int n0 = blockIdx.y * TN;
  v8f acc[2][4] = {};

  stage_tn(Aop, lda, 0, m0, lA[0], scale_a ? scale : nullptr, tid);
  stage_tn(Bop, ldb, 0, n0, lB[0], scale_b ? scale : nullptr, tid);
  int buf = 0;

  for (int k0 = 0; k0 < Kdim; k0 += TK) {
    __syncthreads();
    if (k0 + TK < Kdim) {
      stage_tn(Aop, lda, k0 + TK, m0, lA[buf ^ 1], scale_a ? scale : nullptr, tid);
      stage_tn(Bop, ldb, k0 + TK, n0, lB[buf ^ 1], scale_b ? scale : nullptr, tid);
      if (k0 + 2 * TK < Kdim) {
        __builtin_prefetch(Aop + (size_t)(k0 + 2 * TK + (tid >> 5)) * lda + m0, 0, 3);
        __builtin_prefetch(Bop + (size_t)(k0 + 2 * TK + (tid >> 5)) * ldb + n0, 0, 3);
      }
    }
    compute_tn(lA[buf], lB[buf], wm, wn, ncol, half, acc);
    buf ^= 1;
  }
#pragma unroll
  for (int mi = 0; mi < 2; ++mi)
#pragma unroll
    for (int ni = 0; ni < 4; ++ni) {
      int gn = n0 + wn * 64 + ni * 16 + ncol;
#pragma unroll
      for (int v = 0; v < 8; ++v) {
        int gm = m0 + wm * 32 + mi * 16 + v + 8 * half;
        size_t idx = (size_t)gm * ldc + gn;
        out[idx] = base[idx] + alpha * acc[mi][ni][v];
      }
    }
}

// ---------------------------------------------------------------------------
extern "C" void kernel_launch(void* const* d_in, const int* in_sizes, int n_in,
                              void* d_out, int out_size, void* d_ws, size_t ws_size,
                              hipStream_t stream) {
  (void)in_sizes; (void)n_in; (void)out_size; (void)ws_size;
  const float* x   = (const float*)d_in[0];  // [B, I]
  const float* s   = (const float*)d_in[1];  // [B, H]
  const float* y   = (const float*)d_in[2];  // [B, H]
  const float* P   = (const float*)d_in[3];  // [I, I]
  const float* win = (const float*)d_in[4];  // [H, I]
  const float* wr  = (const float*)d_in[5];  // [H, H]

  float* out = (float*)d_out;
  const size_t BH = (size_t)B_DIM * H_DIM;
  float* out_err = out;                                    // [B, H]
  float* out_r   = out + BH;                               // [B, H]
  float* out_sn  = out + 2 * BH;                           // [B, H]
  float* out_P   = out + 3 * BH;                           // [I, I]
  float* out_win = out_P + (size_t)I_DIM * I_DIM;          // [H, I]

  float* ws_kf    = (float*)d_ws;                          // [B, I] f32 (64 MB)
  float* ws_scale = ws_kf + (size_t)B_DIM * I_DIM;         // [B]    f32

  dim3 blk(256);

  // 1) rx + err + r + s_new (fused)
  rx_fused_kernel<<<dim3(B_DIM / TM, H_DIM / TN), blk, 0, stream>>>(
      x, win, s, wr, y, out_err, out_r, out_sn);

  // 2) k_fenmu = x @ P^T
  gemm_nt_kernel<<<dim3(B_DIM / TM, I_DIM / TN), blk, 0, stream>>>(
      x, P, ws_kf, I_DIM, I_DIM, I_DIM, I_DIM);

  // 3) scale[b] = 1/(1 + rPr[b])
  rowscale_kernel<<<dim3(B_DIM), blk, 0, stream>>>(ws_kf, x, ws_scale);

  // 4) win_new = win - (1/B) * err^T @ k        (k = kf * scale, on the fly)
  update_tn_kernel<<<dim3(H_DIM / TM, I_DIM / TN), blk, 0, stream>>>(
      out_err, H_DIM, ws_kf, I_DIM, win, out_win, I_DIM,
      ws_scale, /*scale_a=*/0, /*scale_b=*/1, B_DIM, -1.0f / (float)B_DIM);

  // 5) P_new = P - (1/B) * k^T @ kf
  update_tn_kernel<<<dim3(I_DIM / TM, I_DIM / TN), blk, 0, stream>>>(
      ws_kf, I_DIM, ws_kf, I_DIM, P, out_P, I_DIM,
      ws_scale, /*scale_a=*/1, /*scale_b=*/0, B_DIM, -1.0f / (float)B_DIM);
}
